// MultiHeadAttention_62577673503448
// MI455X (gfx1250) — matrix operand order repaired and checked by reading, not compiled, not run on hardware
//
#include <hip/hip_runtime.h>
#include <hip/hip_fp16.h>
#include <math.h>

// ---------------------------------------------------------------------------
// MultiHeadAttention (B=4,T=2048,D=512,H=4,dk=128) for gfx1250 / MI455X.
// GEMMs via v_wmma_f32_16x16x32_f16 (wave32).  Attention stages K/V chunks
// into LDS with gfx1250 async-to-LDS copies (ASYNCcnt, double buffered) so
// the 8 waves of a block share each K/V read (8x L2 traffic reduction).
// ---------------------------------------------------------------------------

typedef __attribute__((ext_vector_type(16))) _Float16 v16h;
typedef __attribute__((ext_vector_type(8)))  _Float16 v8h;
typedef __attribute__((ext_vector_type(8)))  float    v8f;

#define WMMA_F16(a, b, c) \
  __builtin_amdgcn_wmma_f32_16x16x32_f16(false, (a), false, (b), (short)0, (c), false, false)

constexpr int B_  = 4;
constexpr int T_  = 2048;
constexpr int D_  = 512;
constexpr int H_  = 4;
constexpr int DK_ = 128;
constexpr int BT_ = B_ * T_;

// 16-byte global -> LDS async copy (gfx1250).  VDST = per-lane LDS byte
// address (low 32 bits of the flat shared pointer), VADDR = global address.
__device__ __forceinline__ void async_copy_b128(uint32_t lds_off, const void* gaddr) {
  asm volatile("global_load_async_to_lds_b128 %0, %1, off"
               :: "v"(lds_off), "v"(gaddr) : "memory");
}

// ---------------------------------------------------------------------------
// Projection: Y = relu(X @ W^T + bias).  One wave per 32x32 output tile
// (2 A-frags x 2 B-frags, 4 WMMAs per 32-wide K step) to halve X and W
// re-read traffic vs a 16x16 tile.
// mode 0: f16 row-major; 1: f16 into Vt[b][h][dk][T]; 2: f32 (residual R).
// ---------------------------------------------------------------------------
__global__ __launch_bounds__(256)
void proj_kernel(const float* __restrict__ X, const float* __restrict__ W,
                 const float* __restrict__ bias,
                 _Float16* __restrict__ Yh, float* __restrict__ Yf, int mode)
{
  const int lane = threadIdx.x & 31;
  const int wave = threadIdx.x >> 5;
  const int w    = blockIdx.x * 8 + wave;   // 4096 waves total
  const int t0   = (w >> 4) * 32;           // row tile in [0, BT)
  const int o0   = (w & 15) * 32;           // col tile in [0, D)
  const int lm   = lane & 15;
  const int lh   = lane >> 4;

  const float* xr[2] = { X + (size_t)(t0 + lm) * D_, X + (size_t)(t0 + 16 + lm) * D_ };
  const float* wr[2] = { W + (size_t)(o0 + lm) * D_, W + (size_t)(o0 + 16 + lm) * D_ };

  v8f acc[2][2];
  #pragma unroll
  for (int i = 0; i < 2; ++i)
    #pragma unroll
    for (int j = 0; j < 2; ++j) acc[i][j] = v8f{};

  for (int kb = 0; kb < D_; kb += 32) {
    union { v16h v; _Float16 e[16]; } a[2], b[2];
    #pragma unroll
    for (int i = 0; i < 2; ++i) {
      const float* xa = xr[i] + kb + 16 * lh; // K {0-15}/{16-31}
      #pragma unroll
      for (int k = 0; k < 8; ++k) {
        a[i].e[k]     = (_Float16)xa[k];
        a[i].e[8 + k] = (_Float16)xa[8 + k];
      }
    }
    #pragma unroll
    for (int j = 0; j < 2; ++j) {
      const float* wb = wr[j] + kb + 16 * lh; // K {0-15}/{16-31}
      #pragma unroll
      for (int k = 0; k < 16; ++k) b[j].e[k] = (_Float16)wb[k];
    }
    #pragma unroll
    for (int i = 0; i < 2; ++i)
      #pragma unroll
      for (int j = 0; j < 2; ++j)
        acc[i][j] = WMMA_F16(a[i].v, b[j].v, acc[i][j]);
  }

  #pragma unroll
  for (int j = 0; j < 2; ++j) {
    const float bv = bias[o0 + 16 * j + lm];
    #pragma unroll
    for (int i = 0; i < 2; ++i) {
      float y[8];
      #pragma unroll
      for (int r = 0; r < 8; ++r) {
        float v = acc[i][j][r] + bv;
        y[r] = v > 0.f ? v : 0.f;
      }
      const int tt = t0 + 16 * i;
      const int oo = o0 + 16 * j;
      if (mode == 0) {                           // f16 row-major [BT][D]
        #pragma unroll
        for (int r = 0; r < 8; ++r)
          Yh[(size_t)(tt + r + 8 * lh) * D_ + oo + lm] = (_Float16)y[r];
      } else if (mode == 1) {                    // Vt[b][h][dk][T]
        const int o  = oo + lm;
        const int bb = tt / T_;
        const int t  = (tt % T_) + 8 * lh;       // 8 contiguous t per lane
        union { v8h v; _Float16 e[8]; } st;
        #pragma unroll
        for (int r = 0; r < 8; ++r) st.e[r] = (_Float16)y[r];
        *(v8h*)(Yh + ((size_t)((bb * H_ + o / DK_) * DK_ + o % DK_)) * T_ + t) = st.v;
      } else {                                   // f32 row-major (residual)
        #pragma unroll
        for (int r = 0; r < 8; ++r)
          Yf[(size_t)(tt + r + 8 * lh) * D_ + oo + lm] = y[r];
      }
    }
  }
}

// ---------------------------------------------------------------------------
// Flash attention.  Block = 8 waves on ONE (b,h) with 8 adjacent 16-query
// tiles; each 32-key K chunk (8KB) + V chunk (8KB) is async-copied to LDS
// once per block (double buffered, ASYNCcnt-tracked).  S^T tiles (M=key,
// N=query) land in C layout == P A-fragment layout -> no shuffle transpose.
// ---------------------------------------------------------------------------
constexpr int KROW = 136;   // padded row (halfs): 272B -> fragment reads spread banks
constexpr int VROW = 40;    // padded row (halfs): 80B
constexpr int NCHUNK = T_ / 32;

__global__ __launch_bounds__(256)
void attn_kernel(const _Float16* __restrict__ Qh, const _Float16* __restrict__ Kh,
                 const _Float16* __restrict__ Vt, float* __restrict__ O)
{
  __shared__ __align__(32) _Float16 Kbuf[2][32][KROW];
  __shared__ __align__(32) _Float16 Vbuf[2][128][VROW];

  const int tid  = threadIdx.x;
  const int lane = tid & 31;
  const int wave = tid >> 5;
  const int blk  = blockIdx.x;                 // 256 blocks
  const int h    = (blk >> 4) & 3;
  const int b    = blk >> 6;
  const int q0   = ((blk & 15) * 8 + wave) * 16;
  const int lm   = lane & 15;
  const int lh   = lane >> 4;
  const float scale = 0.08838834764831845f;    // 1/sqrt(128)

  const _Float16* kbase = Kh + (size_t)(b * T_) * D_ + h * DK_;
  const _Float16* vbase = Vt + (size_t)((b * H_ + h) * DK_) * T_;

  // Q B-fragments (resident for the whole key loop)
  v16h qb[4];
  const _Float16* qbase = Qh + (size_t)(b * T_ + q0 + lm) * D_ + h * DK_;
  #pragma unroll
  for (int d = 0; d < 4; ++d)
    qb[d] = *(const v16h*)(qbase + d * 32 + 16 * lh);

  // Stage one 32-key chunk: K 32x128 halfs (512 x 16B), V 128x32 halfs.
  // 256 threads x (2+2) async b128 ops -> 4 per wave per chunk.
  auto stage = [&](int c, int p) {
    #pragma unroll
    for (int u = 0; u < 2; ++u) {
      const int j = tid * 2 + u;
      const int row = j >> 4, seg = j & 15;
      async_copy_b128((uint32_t)(uintptr_t)&Kbuf[p][row][seg * 8],
                      kbase + (size_t)(c + row) * D_ + seg * 8);
    }
    #pragma unroll
    for (int u = 0; u < 2; ++u) {
      const int j = tid * 2 + u;
      const int row = j >> 2, seg = j & 3;
      async_copy_b128((uint32_t)(uintptr_t)&Vbuf[p][row][seg * 8],
                      vbase + (size_t)row * T_ + c + seg * 8);
    }
  };

  v8f o[8];
  #pragma unroll
  for (int j = 0; j < 8; ++j) o[j] = v8f{};
  float m = -INFINITY, l = 0.f;

  stage(0, 0);
  for (int ci = 0; ci < NCHUNK; ++ci) {
    const int p = ci & 1;
    if (ci + 1 < NCHUNK) {
      stage((ci + 1) * 32, p ^ 1);             // buffer p^1 free since iter ci-1
      asm volatile("s_wait_asynccnt 0x4" ::: "memory");  // chunk ci done (in-order)
    } else {
      asm volatile("s_wait_asynccnt 0x0" ::: "memory");
    }
    __syncthreads();                           // all waves' copies visible

    // --- S^T = K * Q^T for 32 keys x 16 queries (from LDS) ---
    v8f s0 = {}, s1 = {};
    #pragma unroll
    for (int d = 0; d < 4; ++d) {
      union { v16h v; v8h h2[2]; } a0, a1;
      const _Float16* kp0 = &Kbuf[p][lm][d * 32 + 16 * lh];
      const _Float16* kp1 = &Kbuf[p][16 + lm][d * 32 + 16 * lh];
      a0.h2[0] = *(const v8h*)kp0;
      a0.h2[1] = *(const v8h*)(kp0 + 8);
      a1.h2[0] = *(const v8h*)kp1;
      a1.h2[1] = *(const v8h*)(kp1 + 8);
      s0 = WMMA_F16(a0.v, qb[d], s0);
      s1 = WMMA_F16(a1.v, qb[d], s1);
    }

    // --- online softmax; every value in this lane belongs to query lm ---
    float f0[8], f1[8];
    float cm = -INFINITY;
    #pragma unroll
    for (int r = 0; r < 8; ++r) {
      f0[r] = s0[r] * scale;
      f1[r] = s1[r] * scale;
      cm = fmaxf(cm, fmaxf(f0[r], f1[r]));
    }
    cm = fmaxf(cm, __shfl_xor(cm, 16, 32));    // combine lane pair (L, L^16)
    const float mn = fmaxf(m, cm);

    union { v16h v; _Float16 e[16]; } pa;      // P A-fragment (layout match)
    float cs = 0.f;
    #pragma unroll
    for (int r = 0; r < 8; ++r) {
      float p0 = __expf(f0[r] - mn);
      float p1 = __expf(f1[r] - mn);
      cs += p0 + p1;
      pa.e[r]     = (_Float16)p0;
      pa.e[8 + r] = (_Float16)p1;
    }
    cs += __shfl_xor(cs, 16, 32);
    const float fac = __expf(m - mn);
    l = l * fac + cs;
    m = mn;

    float fo[8];                               // factor for query r + 8*lh
    #pragma unroll
    for (int r = 0; r < 8; ++r) fo[r] = __shfl(fac, r + 8 * lh, 32);
    #pragma unroll
    for (int j = 0; j < 8; ++j)
      #pragma unroll
      for (int r = 0; r < 8; ++r) o[j][r] *= fo[r];

    // --- O += P @ V ; B-frags from padded LDS rows ---
    #pragma unroll
    for (int j = 0; j < 8; ++j) {
      union { v16h v; v8h h2[2]; } vb;
      const _Float16* vp = &Vbuf[p][j * 16 + lm][8 * lh];
      vb.h2[0] = *(const v8h*)vp;
      vb.h2[1] = *(const v8h*)(vp + 16);
      o[j] = WMMA_F16(pa.v, vb.v, o[j]);
    }
    __syncthreads();                           // readers done before restage
  }

  float linv[8];
  #pragma unroll
  for (int r = 0; r < 8; ++r) linv[r] = 1.f / __shfl(l, r + 8 * lh, 32);

  float* obase = O + (size_t)(b * T_ + q0) * D_ + h * DK_;
  #pragma unroll
  for (int j = 0; j < 8; ++j)
    #pragma unroll
    for (int r = 0; r < 8; ++r)
      obase[(size_t)(r + 8 * lh) * D_ + j * 16 + lm] = o[j][r] * linv[r];
}

// ---------------------------------------------------------------------------
// Residual add + LayerNorm (unbiased std, eps added to std).
// ---------------------------------------------------------------------------
__global__ __launch_bounds__(256)
void ln_kernel(const float* __restrict__ O, const float* __restrict__ R,
               const float* __restrict__ gamma, const float* __restrict__ beta,
               float* __restrict__ out)
{
  __shared__ float red[256];
  const int row = blockIdx.x;
  const int tid = threadIdx.x;
  const float* orow = O + (size_t)row * D_;
  const float* rrow = R + (size_t)row * D_;

  const float x0 = orow[tid] + rrow[tid];
  const float x1 = orow[tid + 256] + rrow[tid + 256];

  red[tid] = x0 + x1;
  __syncthreads();
  for (int off = 128; off > 0; off >>= 1) {
    if (tid < off) red[tid] += red[tid + off];
    __syncthreads();
  }
  const float mean = red[0] * (1.f / 512.f);
  __syncthreads();

  const float d0 = x0 - mean, d1 = x1 - mean;
  red[tid] = d0 * d0 + d1 * d1;
  __syncthreads();
  for (int off = 128; off > 0; off >>= 1) {
    if (tid < off) red[tid] += red[tid + off];
    __syncthreads();
  }
  const float var = red[0] * (1.f / 511.f);     // ddof=1
  const float inv = 1.f / (sqrtf(var) + 1e-8f); // eps added to std

  out[(size_t)row * D_ + tid]       = gamma[tid] * d0 * inv + beta[tid];
  out[(size_t)row * D_ + tid + 256] = gamma[tid + 256] * d1 * inv + beta[tid + 256];
}

// ---------------------------------------------------------------------------
extern "C" void kernel_launch(void* const* d_in, const int* in_sizes, int n_in,
                              void* d_out, int out_size, void* d_ws, size_t ws_size,
                              hipStream_t stream)
{
  const float* queries = (const float*)d_in[0];
  const float* keys    = (const float*)d_in[1];
  const float* Wq      = (const float*)d_in[2];
  const float* bq      = (const float*)d_in[3];
  const float* Wk      = (const float*)d_in[4];
  const float* bk      = (const float*)d_in[5];
  const float* Wv      = (const float*)d_in[6];
  const float* bv      = (const float*)d_in[7];
  const float* Wr      = (const float*)d_in[8];
  const float* br      = (const float*)d_in[9];
  const float* gamma   = (const float*)d_in[10];
  const float* beta    = (const float*)d_in[11];
  float* out           = (float*)d_out;

  // Workspace carve-up (56 MB total)
  char* ws = (char*)d_ws;
  _Float16* Qh = (_Float16*)(ws);
  _Float16* Kh = (_Float16*)(ws + (size_t)BT_ * D_ * 2);
  _Float16* Vt = (_Float16*)(ws + (size_t)BT_ * D_ * 4);
  float*    Rf = (float*)   (ws + (size_t)BT_ * D_ * 6);
  float*    Of = (float*)   (ws + (size_t)BT_ * D_ * 6 + (size_t)BT_ * D_ * 4);

  const dim3 blk(256);
  // 4096 32x32 tiles / 8 waves per block
  proj_kernel<<<512, blk, 0, stream>>>(queries, Wq, bq, Qh, nullptr, 0);
  proj_kernel<<<512, blk, 0, stream>>>(keys,    Wk, bk, Kh, nullptr, 0);
  proj_kernel<<<512, blk, 0, stream>>>(keys,    Wv, bv, Vt, nullptr, 1);
  proj_kernel<<<512, blk, 0, stream>>>(queries, Wr, br, nullptr, Rf, 2);

  attn_kernel<<<256, blk, 0, stream>>>(Qh, Kh, Vt, Of);

  ln_kernel<<<BT_, blk, 0, stream>>>(Of, Rf, gamma, beta, out);
}
